// Block_38689065402733
// MI455X (gfx1250) — matrix operand order repaired and checked by reading, not compiled, not run on hardware
//
#include <hip/hip_runtime.h>
#include <hip/hip_bf16.h>
#include <math.h>

#define B_  2
#define T_  2048
#define D_  1024
#define H_  16
#define HD_ 64
#define FF_ 4096
#define M_  (B_ * T_)

typedef __attribute__((ext_vector_type(16))) __bf16 v16bf;
typedef __attribute__((ext_vector_type(8)))  float  v8f;

union AFrag { v16bf v; uint4 q[2]; };
union PFrag { v16bf v; __bf16 e[16]; };

// B-fragment (32x16 bf16) from transposed [N][K] storage: 16 contiguous bf16.
__device__ __forceinline__ v16bf load_b_frag(const __bf16* base) {
  AFrag f;
  f.q[0] = *reinterpret_cast<const uint4*>(base);
  f.q[1] = *reinterpret_cast<const uint4*>(base + 8);
  return f.v;
}

// Async 16-byte global -> LDS copy (per-lane), tracked by ASYNCcnt.
// LDS byte address = low 32 bits of the generic pointer (LDS aperture).
__device__ __forceinline__ void async_copy16(const void* g, void* l) {
  const unsigned lofs = (unsigned)(size_t)l;
  asm volatile("global_load_async_to_lds_b128 %0, %1, off"
               :: "v"(lofs), "v"((unsigned long long)(size_t)g)
               : "memory");
}
#define WAIT_ASYNC(n) asm volatile("s_wait_asynccnt %0" :: "n"(n) : "memory")

// ---------------------------------------------------------------------------
// GEMM: C[M,N](f32 acc) = A[M,K](bf16,row-major) * Bt[N,K](bf16, B transposed)
//       + bias[N] (+ gelu) (+ residual[M,N]); out to f32 and/or bf16.
// Block = 128 thr = 4 waves -> 64x128 C tile; wave tile 32x64 (2x4 WMMAs).
// A staged to LDS in 64-deep K slices via double-buffered
// global_load_async_to_lds_b128 (16 WMMAs per barrier pair);
// B (weights) read directly from global (L2-resident, 192 MB).
// ---------------------------------------------------------------------------
#define A_PITCH 40   // 32 bf16 + 8 pad -> 80B row pitch, conflict-free b128 reads

__global__ void __launch_bounds__(128)
gemm_bf16_wmma(const __bf16* __restrict__ A, const __bf16* __restrict__ Bt,
               const float* __restrict__ bias, const float* __restrict__ resid,
               float* __restrict__ outF, __bf16* __restrict__ outB,
               int M, int N, int K, int act)
{
  // [buffer][k-subslice][64 rows * A_PITCH]
  __shared__ __align__(16) __bf16 Abuf[2][2][64 * A_PITCH];
  const int tid  = threadIdx.x;
  const int lane = tid & 31;
  const int wv   = tid >> 5;
  const int lm   = lane & 15;
  const int lh   = lane >> 4;
  const int mtile = blockIdx.y * 64;
  const int ntile = blockIdx.x * 128;
  const int wm = (wv >> 1) * 32;   // wave M offset within block tile
  const int wn = (wv & 1) * 64;    // wave N offset within block tile

  v8f acc[2][4] = {};

  const __bf16* bp[4];
#pragma unroll
  for (int j = 0; j < 4; ++j)
    bp[j] = Bt + (size_t)(ntile + wn + j * 16 + lm) * K + lh * 16;

  // stage one 64x64 A slice (k0..k0+63) into Abuf[buf]: 512 x 16B chunks
  auto stage = [&](int buf, int k0) {
#pragma unroll
    for (int i = 0; i < 4; ++i) {
      const int chunk = i * 128 + tid;     // 0..511
      const int slice = chunk >> 8;        // 0..1 (32-wide K sub-slice)
      const int rem   = chunk & 255;
      const int row   = rem >> 2;          // 4 chunks per 64B row
      const int col   = (rem & 3) * 8;
      async_copy16(A + (size_t)(mtile + row) * K + k0 + slice * 32 + col,
                   &Abuf[buf][slice][row * A_PITCH + col]);
    }
  };

  stage(0, 0);
  int p = 0;
  for (int k0 = 0; k0 < K; k0 += 64, p ^= 1) {
    if (k0 + 64 < K) {
      stage(p ^ 1, k0 + 64);
      WAIT_ASYNC(4);               // in-order: current buffer's 4 ops done
    } else {
      WAIT_ASYNC(0);
    }
    __syncthreads();

#pragma unroll
    for (int s = 0; s < 2; ++s) {  // two 32-wide K sub-steps per stage
      v16bf a[2], b[4];
#pragma unroll
      for (int i = 0; i < 2; ++i) {
        const __bf16* ar = &Abuf[p][s][(wm + i * 16 + lm) * A_PITCH + lh * 16];
        AFrag f;
        f.q[0] = *reinterpret_cast<const uint4*>(ar);
        f.q[1] = *reinterpret_cast<const uint4*>(ar + 8);
        a[i] = f.v;
      }
#pragma unroll
      for (int j = 0; j < 4; ++j) b[j] = load_b_frag(bp[j] + k0 + s * 32);
#pragma unroll
      for (int i = 0; i < 2; ++i)
#pragma unroll
        for (int j = 0; j < 4; ++j)
          acc[i][j] = __builtin_amdgcn_wmma_f32_16x16x32_bf16(
              false, a[i], false, b[j], (short)0, acc[i][j], false, false);
    }
    __syncthreads();
  }

  auto emit = [&](const v8f& a8, int mi, int nj) {
    const int n = ntile + wn + nj * 16 + lm;
    const float bvv = bias ? bias[n] : 0.f;
#pragma unroll
    for (int r = 0; r < 8; ++r) {
      const int m = mtile + wm + mi * 16 + lh * 8 + r;
      float val = a8[r] + bvv;
      if (act == 1) val = 0.5f * val * (1.f + erff(val * 0.70710678118654752f));
      const size_t idx = (size_t)m * N + n;
      if (resid) val += resid[idx];
      if (outF) outF[idx] = val;
      if (outB) outB[idx] = (__bf16)val;
    }
  };
#pragma unroll
  for (int i = 0; i < 2; ++i)
#pragma unroll
    for (int j = 0; j < 4; ++j) emit(acc[i][j], i, j);
}

// ---------------------------------------------------------------------------
// LayerNorm: one block per row of D_=1024, write bf16.
// ---------------------------------------------------------------------------
__global__ void __launch_bounds__(256)
layernorm_bf16(const float* __restrict__ x, const float* __restrict__ g,
               const float* __restrict__ bta, __bf16* __restrict__ out)
{
  __shared__ float s1[256], s2[256];
  const int row = blockIdx.x;
  const int tid = threadIdx.x;
  const float* xr = x + (size_t)row * D_;
  float a = 0.f, b2 = 0.f;
  for (int i = tid; i < D_; i += 256) { float v = xr[i]; a += v; b2 += v * v; }
  s1[tid] = a; s2[tid] = b2;
  __syncthreads();
  for (int off = 128; off > 0; off >>= 1) {
    if (tid < off) { s1[tid] += s1[tid + off]; s2[tid] += s2[tid + off]; }
    __syncthreads();
  }
  const float mu  = s1[0] * (1.f / D_);
  const float var = s2[0] * (1.f / D_) - mu * mu;
  const float inv = rsqrtf(var + 1e-5f);
  __bf16* orow = out + (size_t)row * D_;
  for (int i = tid; i < D_; i += 256)
    orow[i] = (__bf16)((xr[i] - mu) * inv * g[i] + bta[i]);
}

// ---------------------------------------------------------------------------
// Weight convert+transpose: wt[n*K + k] = (bf16) w[k*N + n]
// ---------------------------------------------------------------------------
__global__ void __launch_bounds__(256)
transpose_to_bf16(const float* __restrict__ w, __bf16* __restrict__ wt, int K, int N)
{
  const size_t idx = (size_t)blockIdx.x * 256 + threadIdx.x;
  if (idx >= (size_t)K * N) return;
  const int k = (int)(idx % K);
  const int n = (int)(idx / K);
  wt[idx] = (__bf16)w[(size_t)k * N + n];
}

// ---------------------------------------------------------------------------
// Per-head V transpose: vt[((b*H+h)*HD + hd)*T + t] = v[(b*T+t)*D + h*HD + hd]
// ---------------------------------------------------------------------------
__global__ void __launch_bounds__(256)
v_head_transpose(const __bf16* __restrict__ v, __bf16* __restrict__ vt)
{
  const size_t idx = (size_t)blockIdx.x * 256 + threadIdx.x; // B*H*HD*T
  const int t  = (int)(idx & (T_ - 1));
  const size_t rest = idx >> 11;            // / T_
  const int hd = (int)(rest & (HD_ - 1));
  const int bh = (int)(rest >> 6);          // / HD_
  const int h  = bh & (H_ - 1);
  const int b  = bh >> 4;                   // / H_
  vt[idx] = v[(size_t)(b * T_ + t) * D_ + h * HD_ + hd];
}

// ---------------------------------------------------------------------------
// Causal flash attention. One wave per (b, h, 16-row Q tile). 32 keys/step:
//   4 score WMMAs (QK^T over HD=64) -> scale+mask -> LDS stage -> online
//   softmax (stats in registers, pair-combine via shfl_xor 16) -> P (bf16
//   A-fragment) -> 4 PV WMMAs into 16x64 f32 accumulators.
// ---------------------------------------------------------------------------
__global__ void __launch_bounds__(128)
flash_attn_causal(const __bf16* __restrict__ Q, const __bf16* __restrict__ Km,
                  const __bf16* __restrict__ Vt, __bf16* __restrict__ O)
{
  __shared__ float Sld[4][16][33];          // per-wave 16x32 score tile (+pad)
  const int lane = threadIdx.x & 31;
  const int wv   = threadIdx.x >> 5;
  const int lm   = lane & 15;
  const int lh   = lane >> 4;
  const int wid  = blockIdx.x * 4 + wv;
  const int qt   = wid & (T_ / 16 - 1);
  const int bh   = wid / (T_ / 16);
  const int h    = bh & (H_ - 1);
  const int b    = bh / H_;
  const int qbase = qt * 16;

  // Q A-fragments for hd 0..31 and 32..63 (loaded once)
  const __bf16* qrow = Q + (size_t)(b * T_ + qbase + lm) * D_ + h * HD_;
  v16bf qa0, qa1;
  {
    AFrag f;
    f.q[0] = *reinterpret_cast<const uint4*>(qrow + lh * 16);
    f.q[1] = *reinterpret_cast<const uint4*>(qrow + 8 + lh * 16);
    qa0 = f.v;
    f.q[0] = *reinterpret_cast<const uint4*>(qrow + 32 + lh * 16);
    f.q[1] = *reinterpret_cast<const uint4*>(qrow + 40 + lh * 16);
    qa1 = f.v;
  }

  const __bf16* krows = Km + (size_t)(b * T_) * D_ + h * HD_;        // [T][HD]
  const __bf16* vbase = Vt + (size_t)((b * H_ + h) * HD_) * T_;      // [HD][T]

  v8f o0 = {}, o1 = {}, o2 = {}, o3 = {};
  float m_run = -1e30f, l_run = 0.f;

  const int nblocks = (qbase + 15) / 32 + 1;
  for (int jj = 0; jj < nblocks; ++jj) {
    const int kb = jj * 32;
#pragma unroll
    for (int half = 0; half < 2; ++half) {
      // K rows are natural Bt layout: [key][hd] contiguous in hd
      const __bf16* kp = krows + (size_t)(kb + half * 16 + lm) * D_ + lh * 16;
      v16bf k0f = load_b_frag(kp);
      v16bf k1f = load_b_frag(kp + 32);
      v8f s = {};
      s = __builtin_amdgcn_wmma_f32_16x16x32_bf16(false, qa0, false, k0f, (short)0, s, false, false);
      s = __builtin_amdgcn_wmma_f32_16x16x32_bf16(false, qa1, false, k1f, (short)0, s, false, false);
      const int kpos = kb + half * 16 + lm;
#pragma unroll
      for (int r = 0; r < 8; ++r) {
        const int qpos = qbase + lh * 8 + r;
        float val = s[r] * 0.125f;                 // 1/sqrt(HD)
        if (kpos > qpos) val = -1e30f;             // causal mask
        Sld[wv][lh * 8 + r][half * 16 + lm] = val;
      }
    }
    asm volatile("s_wait_dscnt 0x0" ::: "memory"); // wave-local LDS transpose

    // lane owns stats for row lm; each half-wave covers 16 keys
    float tmax = -1e30f;
#pragma unroll
    for (int i = 0; i < 16; ++i) tmax = fmaxf(tmax, Sld[wv][lm][lh * 16 + i]);
    tmax = fmaxf(tmax, __shfl_xor(tmax, 16, 32));
    const float m_new = fmaxf(m_run, tmax);
    float tsum = 0.f;
#pragma unroll
    for (int i = 0; i < 16; ++i) tsum += __expf(Sld[wv][lm][lh * 16 + i] - m_new);
    tsum += __shfl_xor(tsum, 16, 32);
    const float resc = __expf(m_run - m_new);
    l_run = l_run * resc + tsum;
    m_run = m_new;

    // rescale accumulators: acc element r belongs to row lh*8+r
#pragma unroll
    for (int r = 0; r < 8; ++r) {
      const float rs = __shfl(resc, lh * 8 + r, 32);
      o0[r] *= rs; o1[r] *= rs; o2[r] *= rs; o3[r] *= rs;
    }

    // Build P as bf16 A-fragment (row lm, keys per 16x32 A layout)
    PFrag pf;
#pragma unroll
    for (int i = 0; i < 8; ++i) {
      pf.e[i]     = (__bf16)__expf(Sld[wv][lm][lh * 16 + i]      - m_new);
      pf.e[i + 8] = (__bf16)__expf(Sld[wv][lm][lh * 16 + 8 + i] - m_new);
    }

    // O += P * V ; Vt rows give contiguous keys at fixed hd
    const __bf16* vp = vbase + (size_t)lm * T_ + kb + lh * 16;
    o0 = __builtin_amdgcn_wmma_f32_16x16x32_bf16(false, pf.v, false, load_b_frag(vp),            (short)0, o0, false, false);
    o1 = __builtin_amdgcn_wmma_f32_16x16x32_bf16(false, pf.v, false, load_b_frag(vp + 16 * T_), (short)0, o1, false, false);
    o2 = __builtin_amdgcn_wmma_f32_16x16x32_bf16(false, pf.v, false, load_b_frag(vp + 32 * T_), (short)0, o2, false, false);
    o3 = __builtin_amdgcn_wmma_f32_16x16x32_bf16(false, pf.v, false, load_b_frag(vp + 48 * T_), (short)0, o3, false, false);
  }

  const float linv = 1.f / l_run;
  __bf16* obase = O + (size_t)(b * T_ + qbase) * D_ + h * HD_;
#pragma unroll
  for (int r = 0; r < 8; ++r) {
    const float sc = __shfl(linv, lh * 8 + r, 32);
    __bf16* op = obase + (size_t)(lh * 8 + r) * D_ + lm;
    op[0]  = (__bf16)(o0[r] * sc);
    op[16] = (__bf16)(o1[r] * sc);
    op[32] = (__bf16)(o2[r] * sc);
    op[48] = (__bf16)(o3[r] * sc);
  }
}

// ---------------------------------------------------------------------------
extern "C" void kernel_launch(void* const* d_in, const int* in_sizes, int n_in,
                              void* d_out, int out_size, void* d_ws, size_t ws_size,
                              hipStream_t stream)
{
  const float* x   = (const float*)d_in[0];
  // d_in[1], d_in[2]: pad/causal masks (pad is all-false, causal computed analytically)
  const float* g1  = (const float*)d_in[3];
  const float* be1 = (const float*)d_in[4];
  const float* g2  = (const float*)d_in[5];
  const float* be2 = (const float*)d_in[6];
  const float* wq  = (const float*)d_in[7];
  const float* bq  = (const float*)d_in[8];
  const float* wk  = (const float*)d_in[9];
  const float* bk  = (const float*)d_in[10];
  const float* wvw = (const float*)d_in[11];
  const float* bvv = (const float*)d_in[12];
  const float* wo  = (const float*)d_in[13];
  const float* bo  = (const float*)d_in[14];
  const float* w1  = (const float*)d_in[15];
  const float* b1  = (const float*)d_in[16];
  const float* w2  = (const float*)d_in[17];
  const float* b2  = (const float*)d_in[18];

  char* ws = (char*)d_ws;
  const size_t MB = 1ull << 20;
  __bf16* wqt = (__bf16*)(ws + 0 * MB);    // D*D   bf16 (2 MB)
  __bf16* wkt = (__bf16*)(ws + 2 * MB);
  __bf16* wvt = (__bf16*)(ws + 4 * MB);
  __bf16* wot = (__bf16*)(ws + 6 * MB);
  __bf16* w1t = (__bf16*)(ws + 8 * MB);    // D*FF  bf16 (8 MB)
  __bf16* w2t = (__bf16*)(ws + 16 * MB);   // FF*D  bf16 (8 MB)
  __bf16* lnb = (__bf16*)(ws + 24 * MB);   // M*D   bf16 (8 MB), reused for LN2
  __bf16* qb  = (__bf16*)(ws + 32 * MB);   // M*D   bf16
  __bf16* kbf = (__bf16*)(ws + 40 * MB);
  __bf16* vbf = (__bf16*)(ws + 48 * MB);
  __bf16* vtb = (__bf16*)(ws + 56 * MB);
  __bf16* h1  = (__bf16*)(ws + 32 * MB);   // M*FF bf16 (32 MB) reuses q..vt
  __bf16* att = (__bf16*)(ws + 64 * MB);   // M*D bf16
  float*  x1  = (float*)(ws + 72 * MB);    // M*D f32 (16 MB)

  // Weight convert + transpose to [N][K] bf16
  transpose_to_bf16<<<(D_ * D_) / 256, 256, 0, stream>>>(wq,  wqt, D_, D_);
  transpose_to_bf16<<<(D_ * D_) / 256, 256, 0, stream>>>(wk,  wkt, D_, D_);
  transpose_to_bf16<<<(D_ * D_) / 256, 256, 0, stream>>>(wvw, wvt, D_, D_);
  transpose_to_bf16<<<(D_ * D_) / 256, 256, 0, stream>>>(wo,  wot, D_, D_);
  transpose_to_bf16<<<(D_ * FF_) / 256, 256, 0, stream>>>(w1, w1t, D_, FF_);
  transpose_to_bf16<<<(D_ * FF_) / 256, 256, 0, stream>>>(w2, w2t, FF_, D_);

  // LN1
  layernorm_bf16<<<M_, 256, 0, stream>>>(x, g1, be1, lnb);

  // Q, K, V projections
  dim3 blk(128);
  dim3 gD(D_ / 128, M_ / 64);
  gemm_bf16_wmma<<<gD, blk, 0, stream>>>(lnb, wqt, bq,  nullptr, nullptr, qb,  M_, D_, D_, 0);
  gemm_bf16_wmma<<<gD, blk, 0, stream>>>(lnb, wkt, bk,  nullptr, nullptr, kbf, M_, D_, D_, 0);
  gemm_bf16_wmma<<<gD, blk, 0, stream>>>(lnb, wvt, bvv, nullptr, nullptr, vbf, M_, D_, D_, 0);

  // V per-head transpose for PV WMMA B-fragments
  v_head_transpose<<<(B_ * H_ * HD_ * T_) / 256, 256, 0, stream>>>(vbf, vtb);

  // Causal flash attention
  flash_attn_causal<<<(B_ * H_ * (T_ / 16)) / 4, 128, 0, stream>>>(qb, kbf, vtb, att);

  // Output projection + residual: x1 = x + att @ wo + bo
  gemm_bf16_wmma<<<gD, blk, 0, stream>>>(att, wot, bo, x, x1, nullptr, M_, D_, D_, 0);

  // LN2
  layernorm_bf16<<<M_, 256, 0, stream>>>(x1, g2, be2, lnb);

  // FFN1 with exact GELU
  dim3 gF(FF_ / 128, M_ / 64);
  gemm_bf16_wmma<<<gF, blk, 0, stream>>>(lnb, w1t, b1, nullptr, nullptr, h1, M_, FF_, D_, 1);

  // FFN2 + residual -> d_out (f32)
  gemm_bf16_wmma<<<gD, blk, 0, stream>>>(h1, w2t, b2, x1, (float*)d_out, nullptr, M_, D_, FF_, 0);
}
